// Attention_module_56504589746684
// MI455X (gfx1250) — matrix operand order repaired and checked by reading, not compiled, not run on hardware
//
#include <hip/hip_runtime.h>
#include <hip/hip_fp16.h>

typedef __attribute__((ext_vector_type(16))) _Float16 v16h;
typedef __attribute__((ext_vector_type(8)))  float    v8f;

#define B_      2
#define H_      16
#define S_      2048
#define D_      128
#define TILE_M  16

// 256 threads = 8 wave32 per workgroup. One workgroup handles a 16-row query
// tile of one (b,h). LDS: 16 x 2048 f32 score strip (128 KB, dynamic).
__global__ __launch_bounds__(256)
void fa_fwd_kernel(const float* __restrict__ Q,
                   const float* __restrict__ K,
                   const float* __restrict__ V,
                   float* __restrict__ ctx,
                   float* __restrict__ attn)
{
    extern __shared__ float sm[];          // [TILE_M][S_]

    const int bx   = blockIdx.x;           // query tile 0..127
    const int bh   = blockIdx.y;           // 0..31
    const int m0   = bx * TILE_M;
    const int tid  = threadIdx.x;
    const int wave = tid >> 5;
    const int lane = tid & 31;
    const int half = lane >> 4;            // 0: lanes 0-15, 1: lanes 16-31
    const int l16  = lane & 15;
    const float scale = 0.08838834764831845f;   // 1/sqrt(128)

    const size_t qkv_base = (size_t)bh * S_ * D_;

    // ---------------- preload Q A-fragments (D=128 -> 4 k-steps of 32) ------
    // A 16x32 f16 layout: lane<16 row M=l16 holds K {b1..b1+15},
    // b1 = 32*ds + 16*half.
    v16h aq[4];
    {
        const float* qrow = Q + qkv_base + (size_t)(m0 + l16) * D_;
        #pragma unroll
        for (int ds = 0; ds < 4; ++ds) {
            const int b1 = ds * 32 + half * 16;
            #pragma unroll
            for (int i = 0; i < 8; ++i) {
                aq[ds][i]     = (_Float16)qrow[b1 + i];
                aq[ds][i + 8] = (_Float16)qrow[b1 + 8 + i];
            }
        }
    }

    // ---------------- phase 1: S = Q*K^T (causal tiles only) ----------------
    const int numN = bx + 1;                       // key tiles 0..bx
    for (int nt = wave; nt < numN; nt += 8) {
        v8f c = {};
        const float* krow = K + qkv_base + (size_t)(nt * 16 + l16) * D_;
        #pragma unroll
        for (int ds = 0; ds < 4; ++ds) {
            // B 32x16 f16 layout: lane = col n (=key), holds K 16*half..+15
            v16h b;
            const int k0 = ds * 32 + half * 16;
            #pragma unroll
            for (int i = 0; i < 16; ++i) b[i] = (_Float16)krow[k0 + i];
            c = __builtin_amdgcn_wmma_f32_16x16x32_f16(
                    false, aq[ds], false, b, (short)0, c, false, false);
        }
        #pragma unroll
        for (int r = 0; r < 8; ++r) {
            const int M   = r + 8 * half;          // row within tile
            const int col = nt * 16 + l16;         // key index
            float v = c[r] * scale;
            if (col > m0 + M) v = -1.0e30f;        // causal mask
            sm[M * S_ + col] = v;
        }
    }
    __syncthreads();

    // ---------------- phase 2: exact softmax + attention write --------------
    const int klim2 = ((m0 + TILE_M + 31) >> 5) << 5;   // key range for PV
    #pragma unroll
    for (int i = 0; i < 2; ++i) {
        const int r    = wave * 2 + i;             // each wave owns 2 rows
        const int grow = m0 + r;                   // global query row
        float mx = -3.0e38f;
        for (int col = lane; col <= grow; col += 32)
            mx = fmaxf(mx, sm[r * S_ + col]);
        #pragma unroll
        for (int off = 16; off; off >>= 1)
            mx = fmaxf(mx, __shfl_xor(mx, off, 32));
        float sum = 0.0f;
        for (int col = lane; col <= grow; col += 32)
            sum += __expf(sm[r * S_ + col] - mx);
        #pragma unroll
        for (int off = 16; off; off >>= 1)
            sum += __shfl_xor(sum, off, 32);
        const float inv = 1.0f / sum;

        float* arow = attn + ((size_t)bh * S_ + grow) * S_;
        for (int col = lane; col < S_; col += 32) {
            float p = 0.0f;
            if (col <= grow)
                p = __expf(sm[r * S_ + col] - mx) * inv;
            arow[col] = p;                         // full row incl. zeros
            if (col < klim2) sm[r * S_ + col] = p; // normalized P for PV
        }
    }
    __syncthreads();

    // ---------------- phase 3: O = P * V ------------------------------------
    {
        const int dn = wave;                       // 8 waves x 16 cols = D
        v8f c = {};
        for (int kb = 0; kb < klim2; kb += 32) {
            v16h a, b;
            const int b1 = kb + half * 16;
            const float* prow = sm + l16 * S_;     // P row M=l16
            #pragma unroll
            for (int i = 0; i < 8; ++i) {
                a[i]     = (_Float16)prow[b1 + i];
                a[i + 8] = (_Float16)prow[b1 + 8 + i];
            }
            const int k0 = kb + 16 * half;         // key range for this half
            const float* vcol = V + qkv_base + (size_t)k0 * D_ + dn * 16 + l16;
            #pragma unroll
            for (int i = 0; i < 16; ++i)
                b[i] = (_Float16)vcol[(size_t)i * D_];
            c = __builtin_amdgcn_wmma_f32_16x16x32_f16(
                    false, a, false, b, (short)0, c, false, false);
        }
        float* crow = ctx + qkv_base + (size_t)m0 * D_;
        #pragma unroll
        for (int r = 0; r < 8; ++r) {
            const int M = r + 8 * half;
            crow[(size_t)M * D_ + dn * 16 + l16] = c[r];
        }
    }
}

extern "C" void kernel_launch(void* const* d_in, const int* in_sizes, int n_in,
                              void* d_out, int out_size, void* d_ws, size_t ws_size,
                              hipStream_t stream) {
    (void)in_sizes; (void)n_in; (void)out_size; (void)d_ws; (void)ws_size;
    const float* Q = (const float*)d_in[0];
    const float* K = (const float*)d_in[1];
    const float* V = (const float*)d_in[2];
    // d_in[3] = attention_mask (causal, known statically) - unused.

    float* ctx  = (float*)d_out;                               // B*H*S*D
    float* attn = ctx + (size_t)B_ * H_ * S_ * D_;             // B*H*S*S

    const int smem_bytes = TILE_M * S_ * sizeof(float);        // 128 KB
    hipFuncSetAttribute((const void*)fa_fwd_kernel,
                        hipFuncAttributeMaxDynamicSharedMemorySize, smem_bytes);

    dim3 grid(S_ / TILE_M, B_ * H_);   // (128, 32)
    fa_fwd_kernel<<<grid, 256, smem_bytes, stream>>>(Q, K, V, ctx, attn);
}
